// MultiHeadFactorizedDenseAttention_13280038879455
// MI455X (gfx1250) — matrix operand order repaired and checked by reading, not compiled, not run on hardware
//
#include <hip/hip_runtime.h>
#include <stdint.h>

// ---------------------------------------------------------------------------
// MultiHeadFactorizedDenseAttention for MI455X (gfx1250, wave32, WMMA)
//   B=2, S=4096, D=256, H=8, P=R=64, DH=32
//   out = softmax(rowwise outer(l,r)) @ (x@Wv) @ Wo,  l=x@Wl, r=x@Wr
// Flash-style streaming: scores exp(l_p*r_rr - mx) generated on the fly and
// fed to v_wmma_f32_16x16x32_bf16 against LDS-staged V chunks.
// WMMA operands are K-contiguous in LDS -> fragments are 2x ds_load_b128.
// V and weights stored transposed (N-major). Global->LDS staging uses CDNA5
// GLOBAL_LOAD_ASYNC_TO_LDS_B128 (ASYNCcnt) when available, overlapping DMA
// with score generation (attention) / WMMA (pipelined GEMM).
// ---------------------------------------------------------------------------

typedef __bf16 v16bf __attribute__((ext_vector_type(16)));
typedef float  v8f   __attribute__((ext_vector_type(8)));
typedef unsigned short u16;
typedef unsigned int   u32;

#define B_  2
#define S_  4096
#define D_  256
#define M_  (B_ * S_)   // 8192 rows total

#if __has_builtin(__builtin_amdgcn_global_load_async_to_lds_b128) && \
    __has_builtin(__builtin_amdgcn_s_wait_asynccnt)
#define HAVE_ASYNC_LDS 1
#else
#define HAVE_ASYNC_LDS 0
#endif

#if HAVE_ASYNC_LDS
// Builtin prototype (from the compiler diagnostic): payload-typed pointers,
// global (AS1) source and LDS (AS3) destination, plus two immediates.
typedef int i32x4 __attribute__((vector_size(4 * sizeof(int))));
typedef __attribute__((address_space(1))) i32x4 as1_i32x4;
typedef __attribute__((address_space(3))) i32x4 as3_i32x4;
static __device__ __forceinline__ void async_b128(const u16* g, u16* l) {
  __builtin_amdgcn_global_load_async_to_lds_b128((as1_i32x4*)g, (as3_i32x4*)l,
                                                 0, 0);
}
static __device__ __forceinline__ void wait_async0() {
  __builtin_amdgcn_s_wait_asynccnt(0);
}
#endif

struct U4x2 { uint4 a, b; };

static __device__ __forceinline__ u16 f2bf(float f) {
  u32 u = __float_as_uint(f);
  u32 r = u + 0x7FFFu + ((u >> 16) & 1u);   // round-to-nearest-even
  return (u16)(r >> 16);
}

static __device__ __forceinline__ v8f wmma_bf16(v16bf a, v16bf b, v8f c) {
  return __builtin_amdgcn_wmma_f32_16x16x32_bf16(
      /*neg_a=*/false, a, /*neg_b=*/false, b,
      /*c_mod=*/(short)0, c, /*reuse_a=*/false, /*reuse_b=*/false);
}

// A-fragment (16x32 bf16) from row-major LDS tile [M][lda].
// lane%16 = M row; halves 0-7 -> K=kb..kb+7, halves 8-15 -> K=kb+16..kb+23,
// kb = 8*(lane>=16). Two ds_load_b128 per lane. lda multiple of 8.
static __device__ __forceinline__ v16bf load_frag_a(const u16* smem, int lda,
                                                    int row0, int k0) {
  int lane = threadIdx.x & 31;
  const u16* p = smem + (row0 + (lane & 15)) * lda + k0 + ((lane >> 4) << 3);
  U4x2 t;
  t.a = *(const uint4*)(p);
  t.b = *(const uint4*)(p + 16);
  return __builtin_bit_cast(v16bf, t);
}

// B-fragment (32x16 bf16) from COLUMN-MAJOR LDS tile [N][ldt] (K contiguous).
// lane%16 = N col; halves 0-7 -> K=kb..kb+7, halves 8-15 -> K=kb+16..kb+23,
// kb = 8*(lane>=16).
// Two ds_load_b128 per lane. ldt multiple of 8.
static __device__ __forceinline__ v16bf load_frag_bt(const u16* smem, int ldt,
                                                     int k0, int col0) {
  int lane = threadIdx.x & 31;
  const u16* p = smem + (col0 + (lane & 15)) * ldt + k0 + ((lane >> 4) << 3);
  U4x2 t;
  t.a = *(const uint4*)(p);
  t.b = *(const uint4*)(p + 16);
  return __builtin_bit_cast(v16bf, t);
}

// ---------------------------------------------------------------------------
__global__ void k_f32_to_bf16(const float* __restrict__ in,
                              u16* __restrict__ out, int n) {
  int i = blockIdx.x * blockDim.x + threadIdx.x;
  int stride = gridDim.x * blockDim.x;
  for (; i < n; i += stride) out[i] = f2bf(in[i]);
}

// in: Kd x Nd row-major f32;  out: Nd x Kd row-major bf16 (transposed)
__global__ void k_f32_to_bf16_t(const float* __restrict__ in,
                                u16* __restrict__ out, int Kd, int Nd) {
  int i = blockIdx.x * blockDim.x + threadIdx.x;
  int total = Kd * Nd;
  int stride = gridDim.x * blockDim.x;
  for (; i < total; i += stride) {
    int k = i / Nd, n = i - k * Nd;
    out[n * Kd + k] = f2bf(in[i]);
  }
}

// ---------------------------------------------------------------------------
// bf16 WMMA GEMM: C(MxN) = A(MxK, row-major bf16) @ Bt(NxK, N-major bf16)
// 64x64 tile / 256-thread block (8 waves); wave w: rows (w>>1)*16,
// cols (w&1)*32..+32. K multiple of 32. Outputs:
//   WRITE_F32  : Cf row-major MxN f32
//   WRITE_BF16T: Cb transposed NxM bf16 (leading dim M), packed b128 stores
// Async path: double-buffered tiles, DMA overlapped with WMMA.
// ---------------------------------------------------------------------------
template <bool WRITE_F32, bool WRITE_BF16T>
__global__ __launch_bounds__(256) void k_gemm_bf16(
    const u16* __restrict__ A, const u16* __restrict__ Bt,
    float* __restrict__ Cf, u16* __restrict__ Cb, int M, int N, int K) {
  __shared__ u16 Ash[2][64 * 40];   // 64 rows x 32 k, pad to 40
  __shared__ u16 Bsh[2][64 * 40];   // 64 cols x 32 k, pad to 40 (K-contig)

  const int t    = threadIdx.x;
  const int wave = t >> 5;
  const int row0 = blockIdx.x * 64;
  const int col0 = blockIdx.y * 64;
  const int mrow = (wave >> 1) * 16;
  const int ncol = (wave & 1) * 32;
  // tile staging: 256 threads x one 16B vector per operand (64x32 bf16 tile)
  const int sr = t >> 2;            // row (A) / col (Bt) 0..63
  const int sc = (t & 3) * 8;       // k offset 0..24

  v8f acc0 = {};
  v8f acc1 = {};

#if HAVE_ASYNC_LDS
  // ---- software pipeline: async DMA into buf^1, compute from buf ----
  async_b128(&A[(size_t)(row0 + sr) * K + sc], &Ash[0][sr * 40 + sc]);
  async_b128(&Bt[(size_t)(col0 + sr) * K + sc], &Bsh[0][sr * 40 + sc]);
  int buf = 0;
  for (int kk = 0; kk < K; kk += 32, buf ^= 1) {
    wait_async0();        // our async writes into Ash/Bsh[buf] landed
    __syncthreads();      // everyone's landed; prior reads of buf^1 done
    if (kk + 32 < K) {
      async_b128(&A[(size_t)(row0 + sr) * K + kk + 32 + sc],
                 &Ash[buf ^ 1][sr * 40 + sc]);
      async_b128(&Bt[(size_t)(col0 + sr) * K + kk + 32 + sc],
                 &Bsh[buf ^ 1][sr * 40 + sc]);
    }
    v16bf a  = load_frag_a(&Ash[buf][0], 40, mrow, 0);
    v16bf b0 = load_frag_bt(&Bsh[buf][0], 40, 0, ncol);
    v16bf b1 = load_frag_bt(&Bsh[buf][0], 40, 0, ncol + 16);
    acc0 = wmma_bf16(a, b0, acc0);
    acc1 = wmma_bf16(a, b1, acc1);
  }
#else
  for (int kk = 0; kk < K; kk += 32) {
    __syncthreads();
    *(uint4*)(&Ash[0][sr * 40 + sc]) =
        *(const uint4*)(&A[(size_t)(row0 + sr) * K + kk + sc]);
    *(uint4*)(&Bsh[0][sr * 40 + sc]) =
        *(const uint4*)(&Bt[(size_t)(col0 + sr) * K + kk + sc]);
    if (kk + 32 < K) {  // CDNA5 global_prefetch_b8
      __builtin_prefetch(&A[(size_t)(row0 + sr) * K + kk + 32 + sc], 0, 1);
      __builtin_prefetch(&Bt[(size_t)(col0 + sr) * K + kk + 32 + sc], 0, 1);
    }
    __syncthreads();
    v16bf a  = load_frag_a(&Ash[0][0], 40, mrow, 0);
    v16bf b0 = load_frag_bt(&Bsh[0][0], 40, 0, ncol);
    v16bf b1 = load_frag_bt(&Bsh[0][0], 40, 0, ncol + 16);
    acc0 = wmma_bf16(a, b0, acc0);
    acc1 = wmma_bf16(a, b1, acc1);
  }
#endif

  // C layout: lane%16 = N col; VGPR v -> M row = v + 8*(lane>=16).
  int lane  = t & 31;
  int cA    = col0 + ncol + (lane & 15);
  int rbase = row0 + mrow + ((lane >> 4) << 3);
  if (WRITE_F32) {
#pragma unroll
    for (int v = 0; v < 8; ++v) {
      size_t r = (size_t)(rbase + v);
      Cf[r * N + cA]      = acc0[v];
      Cf[r * N + cA + 16] = acc1[v];
    }
  }
  if (WRITE_BF16T) {  // lane owns 8 consecutive M rows of one column: b128
    union { u16 h[8]; uint4 q; } p0, p1;
#pragma unroll
    for (int v = 0; v < 8; ++v) {
      p0.h[v] = f2bf(acc0[v]);
      p1.h[v] = f2bf(acc1[v]);
    }
    *(uint4*)(&Cb[(size_t)cA * M + rbase])        = p0.q;
    *(uint4*)(&Cb[(size_t)(cA + 16) * M + rbase]) = p1.q;
  }
}

// ---------------------------------------------------------------------------
// Attention core. Block = (batch b, 64 query rows). For chunk p (0..63):
//   Sch[n][rr] = exp(l[n,p]*r[n,rr] - mx[n])   (exact row max via extrema)
//   acc[n][d] += Sch @ V[p*64..p*64+63][d]     (bf16 WMMA; V staged from Vt)
// V chunk DMA (async-to-LDS) overlaps with score generation.
// Z accumulates per thread across all 64 chunks; O = acc/Z stored bf16.
// ---------------------------------------------------------------------------
__global__ __launch_bounds__(256) void k_attn(
    const float* __restrict__ lF, const float* __restrict__ rF,
    const u16* __restrict__ Vt,   // 256 x 8192 bf16, d-major
    u16* __restrict__ Obf) {      // 8192 x 256 bf16, row-major
  __shared__ float lT[64 * 64];    // 16 KB
  __shared__ u16   Sch[64 * 72];   //  9 KB  (A operand, row-major)
  __shared__ u16   Vch[256 * 72];  // 36 KB  (B operand, d-major, K-contig)
  __shared__ float mxs[64];
  __shared__ float zs[64];

  const int t     = threadIdx.x;
  const int wave  = t >> 5;
  const int lane  = t & 31;
  const int row0  = blockIdx.x * 64;
  const size_t bS = (size_t)blockIdx.y * S_;
  const int row   = t >> 2;          // score row owned by this thread
  const int cb    = (t & 3) * 16;    // its 16-col slice
  const int mrow  = (wave >> 1) * 16;
  const int dhalf = (wave & 1) * 128;

  // ---- load r slice (regs) and l slice (LDS); compute row extrema ----
  float rr[16], lv16[16];
  {
    const float* rg = rF + (bS + row0 + row) * 64 + cb;
    const float* lg = lF + (bS + row0 + row) * 64 + cb;
#pragma unroll
    for (int q = 0; q < 4; ++q) {
      *(float4*)(&rr[q * 4])   = *(const float4*)(&rg[q * 4]);
      *(float4*)(&lv16[q * 4]) = *(const float4*)(&lg[q * 4]);
      *(float4*)(&lT[row * 64 + cb + q * 4]) = *(const float4*)(&lg[q * 4]);
    }
  }
  float lmx = -3.0e38f, lmn = 3.0e38f, rmx = -3.0e38f, rmn = 3.0e38f;
#pragma unroll
  for (int j = 0; j < 16; ++j) {
    lmx = fmaxf(lmx, lv16[j]); lmn = fminf(lmn, lv16[j]);
    rmx = fmaxf(rmx, rr[j]);   rmn = fminf(rmn, rr[j]);
  }
#pragma unroll
  for (int off = 1; off < 4; off <<= 1) {
    lmx = fmaxf(lmx, __shfl_xor(lmx, off, 32));
    lmn = fminf(lmn, __shfl_xor(lmn, off, 32));
    rmx = fmaxf(rmx, __shfl_xor(rmx, off, 32));
    rmn = fminf(rmn, __shfl_xor(rmn, off, 32));
  }
  if ((t & 3) == 0) {
    float m01 = fmaxf(lmx * rmx, lmx * rmn);
    float m23 = fmaxf(lmn * rmx, lmn * rmn);
    mxs[row] = fmaxf(m01, m23);
  }
  __syncthreads();
  const float mxrow = mxs[row];

  v8f acc[8];
#pragma unroll
  for (int dt = 0; dt < 8; ++dt) acc[dt] = (v8f){};
  float Zpart = 0.f;

  // ------------------------------ main loop ------------------------------
  for (int p = 0; p < 64; ++p) {
    __syncthreads();  // previous chunk fully consumed (dscnt waited @barrier)

    // Stage V chunk from Vt: 256 d-rows x 64 m (128B contiguous per d-row).
    const size_t mbase = bS + (size_t)p * 64;
#pragma unroll
    for (int i = 0; i < 8; ++i) {
      int idx = t + i * 256;          // 2048 x 16B
      int d = idx >> 3, c = (idx & 7) * 8;
#if HAVE_ASYNC_LDS
      async_b128(&Vt[(size_t)d * M_ + mbase + c], &Vch[d * 72 + c]);
#else
      *(uint4*)(&Vch[d * 72 + c]) =
          *(const uint4*)(&Vt[(size_t)d * M_ + mbase + c]);
#endif
    }
#if !HAVE_ASYNC_LDS
    if (p + 1 < 64)  // CDNA5 global_prefetch_b8: next chunk
      __builtin_prefetch(&Vt[(size_t)t * M_ + mbase + 64], 0, 1);
#endif

    // Score chunk: exp(l[row,p]*r[row,rr] - mx[row]) -- overlaps the DMA.
    {
      float lv = lT[row * 64 + p];
#pragma unroll
      for (int j = 0; j < 16; ++j) {
        float s = __expf(fmaf(lv, rr[j], -mxrow));
        Zpart += s;
        Sch[row * 72 + cb + j] = f2bf(s);
      }
    }
#if HAVE_ASYNC_LDS
    wait_async0();   // our V-chunk DMA landed
#endif
    __syncthreads();

    // acc(64x256) += Sch(64x64) @ V(64x256): 2 ksteps x 8 dtiles per wave.
#pragma unroll
    for (int ks = 0; ks < 2; ++ks) {
      v16bf a = load_frag_a(Sch, 72, mrow, ks * 32);
#pragma unroll
      for (int dt = 0; dt < 8; ++dt) {
        v16bf bfrg = load_frag_bt(Vch, 72, ks * 32, dhalf + dt * 16);
        acc[dt] = wmma_bf16(a, bfrg, acc[dt]);
      }
    }
  }

  // ------------------------------ epilogue -------------------------------
  {
    float z = Zpart;
    z += __shfl_xor(z, 1, 32);
    z += __shfl_xor(z, 2, 32);
    if ((t & 3) == 0) zs[row] = z;
  }
  __syncthreads();

  int colb = dhalf + (lane & 15);
  int rloc = mrow + ((lane >> 4) << 3);
#pragma unroll
  for (int v = 0; v < 8; ++v) {
    int rl = rloc + v;
    float zinv = 1.0f / zs[rl];
    size_t base = (bS + row0 + rl) * 256;
#pragma unroll
    for (int dt = 0; dt < 8; ++dt)
      Obf[base + colb + dt * 16] = f2bf(acc[dt][v] * zinv);
  }
}

// ---------------------------------------------------------------------------
extern "C" void kernel_launch(void* const* d_in, const int* in_sizes, int n_in,
                              void* d_out, int out_size, void* d_ws,
                              size_t ws_size, hipStream_t stream) {
  (void)in_sizes; (void)n_in; (void)out_size; (void)ws_size;
  const float* x  = (const float*)d_in[0];
  const float* Wl = (const float*)d_in[1];
  const float* Wr = (const float*)d_in[2];
  const float* Wv = (const float*)d_in[3];
  const float* Wo = (const float*)d_in[4];

  char* ws = (char*)d_ws;
  size_t off = 0;
  auto carve = [&](size_t bytes) {
    void* p = ws + off;
    off += (bytes + 255) & ~(size_t)255;
    return p;
  };
  u16*   xbf = (u16*)  carve((size_t)M_ * D_ * 2);  // 4 MB, row-major MxK
  u16*   Vt  = (u16*)  carve((size_t)M_ * D_ * 2);  // 4 MB, 256x8192 d-major
  u16*   Obf = (u16*)  carve((size_t)M_ * D_ * 2);  // 4 MB, row-major
  float* lF  = (float*)carve((size_t)M_ * 64 * 4);  // 2 MB
  float* rF  = (float*)carve((size_t)M_ * 64 * 4);  // 2 MB
  u16*   Wvt = (u16*)  carve((size_t)D_ * D_ * 2);  // transposed weights NxK
  u16*   Wlt = (u16*)  carve((size_t)D_ * 64 * 2);
  u16*   Wrt = (u16*)  carve((size_t)D_ * 64 * 2);
  u16*   Wot = (u16*)  carve((size_t)D_ * D_ * 2);

  // 1) conversions: x plain; weights converted AND transposed to NxK
  k_f32_to_bf16<<<dim3(1024), dim3(256), 0, stream>>>(x, xbf, M_ * D_);
  k_f32_to_bf16_t<<<dim3(64), dim3(256), 0, stream>>>(Wv, Wvt, D_, D_);
  k_f32_to_bf16_t<<<dim3(16), dim3(256), 0, stream>>>(Wl, Wlt, D_, 64);
  k_f32_to_bf16_t<<<dim3(16), dim3(256), 0, stream>>>(Wr, Wrt, D_, 64);
  k_f32_to_bf16_t<<<dim3(64), dim3(256), 0, stream>>>(Wo, Wot, D_, D_);

  // 2) projections: V -> transposed bf16 (d-major), l/r -> f32
  k_gemm_bf16<false, true><<<dim3(M_ / 64, D_ / 64), dim3(256), 0, stream>>>(
      xbf, Wvt, nullptr, Vt, M_, D_, D_);
  k_gemm_bf16<true, false><<<dim3(M_ / 64, 1), dim3(256), 0, stream>>>(
      xbf, Wlt, lF, nullptr, M_, 64, D_);
  k_gemm_bf16<true, false><<<dim3(M_ / 64, 1), dim3(256), 0, stream>>>(
      xbf, Wrt, rF, nullptr, M_, 64, D_);

  // 3) streaming softmax(outer(l,r)) @ V  -> O (bf16 row-major)
  k_attn<<<dim3(S_ / 64, B_), dim3(256), 0, stream>>>(lF, rF, Vt, Obf);

  // 4) out = O @ Wo (f32)
  k_gemm_bf16<true, false><<<dim3(M_ / 64, D_ / 64), dim3(256), 0, stream>>>(
      Obf, Wot, (float*)d_out, nullptr, M_, D_, D_);
}
